// OutputPPBlock_19112604467445
// MI455X (gfx1250) — matrix operand order repaired and checked by reading, not compiled, not run on hardware
//
#include <hip/hip_runtime.h>
#include <hip/hip_bf16.h>
#include <math.h>

typedef __attribute__((ext_vector_type(16))) __bf16 v16bf;
typedef __attribute__((ext_vector_type(8)))  float  v8f;

union FragAB { v16bf v; uint4 q[2]; };

__device__ __forceinline__ unsigned short f2bf(float x) {
  unsigned u = __float_as_uint(x);
  u += 0x7FFFu + ((u >> 16) & 1u);       // round-to-nearest-even
  return (unsigned short)(u >> 16);
}
__device__ __forceinline__ float bf2f(unsigned short h) {
  return __uint_as_float(((unsigned)h) << 16);
}

constexpr int D_HID = 256;   // OUT
constexpr int EMB   = 128;
constexpr int BM    = 64;    // rows per block (amortizes L2 weight traffic)
constexpr int LDA   = D_HID + 8;  // padded LDS row (bf16 elems), 528B rows keep 16B alignment
constexpr float LN_EPS = 1e-5f;

// ---------------------------------------------------------------------------
// Pack W [256, K] (row-major f32, y = x @ W^T) into bf16 WMMA B-fragments.
// B tile (kt,nt) is 32(K)x16(N): lanes 0-15 hold N=lane for K=kt*32+[0,16),
// lanes 16-31 hold N=lane-16 for K=kt*32+[16,32); pairs (2v,2v+1) per VGPR
// => each lane owns 16 contiguous elements of row n of W. One 32B store/lane.
// ---------------------------------------------------------------------------
__global__ void pack_w_kernel(const float* __restrict__ W,
                              unsigned short* __restrict__ out, int K) {
  const int KT   = K >> 5;
  const int tile = blockIdx.x;            // tile = nt*KT + kt
  const int nt   = tile / KT, kt = tile % KT;
  const int lane = threadIdx.x;
  const int n    = nt * 16 + (lane & 15);
  const int k0   = kt * 32 + ((lane >> 4) << 4);
  const float* src = W + (size_t)n * K + k0;
  unsigned short* dst = out + (((size_t)tile * 32 + lane) << 4);
  #pragma unroll
  for (int i = 0; i < 16; ++i) dst[i] = f2bf(src[i]);
}

__global__ void zero_kernel(float* __restrict__ p, int n) {
  int i = blockIdx.x * blockDim.x + threadIdx.x;
  if (i < n) p[i] = 0.f;
}

// ---------------------------------------------------------------------------
// Node-branch scatter: agg[src[e]] += m[e] * (rbf[e] @ W_rbf_n^T)
// ---------------------------------------------------------------------------
__global__ void scatter_tn_kernel(const float* __restrict__ m,
                                  const float* __restrict__ rbf,
                                  const int* __restrict__ src,
                                  const float* __restrict__ Wrbf,
                                  float* __restrict__ agg, int total) {
  const int idx = blockIdx.x * 256 + threadIdx.x;   // over E*128
  if (idx >= total) return;
  const int e = idx >> 7, f = idx & 127;
  float g = 0.f;
  #pragma unroll
  for (int r = 0; r < 6; ++r) g += rbf[(size_t)e * 6 + r] * Wrbf[f * 6 + r];
  const float v = m[idx] * g;
  __hip_atomic_fetch_add(&agg[(size_t)src[e] * EMB + f], v,
                         __ATOMIC_RELAXED, __HIP_MEMORY_SCOPE_AGENT);
}

// ---------------------------------------------------------------------------
// One wave: 64 rows (4 m-tiles) x 32 cols (2 n-tiles). B fragments reused by
// 4 WMMAs, A fragments by 2. A-fragment (matches packed B): lane<16 -> row=lane,
// K [0,16); lane>=16 -> row=lane-16, K [16,32).
// ---------------------------------------------------------------------------
template<int KT>
__device__ __forceinline__ void gemm_mt4(const unsigned short* Abase, int lane,
                                         const unsigned short* __restrict__ pW,
                                         int nt0, v8f acc[4][2]) {
  #pragma unroll
  for (int kt = 0; kt < KT; ++kt) {
    FragAB b[2];
    #pragma unroll
    for (int j = 0; j < 2; ++j) {
      const uint4* bp = reinterpret_cast<const uint4*>(
          pW + ((((size_t)(nt0 + j) * KT + kt) * 32 + lane) << 4));
      b[j].q[0] = bp[0];
      b[j].q[1] = bp[1];
    }
    #pragma unroll
    for (int mt = 0; mt < 4; ++mt) {
      FragAB a;
      const uint4* ap = reinterpret_cast<const uint4*>(
          Abase + (mt * 16 + (lane & 15)) * LDA + kt * 32 + ((lane >> 4) << 4));
      a.q[0] = ap[0];
      a.q[1] = ap[1];
      #pragma unroll
      for (int j = 0; j < 2; ++j)
        acc[mt][j] = __builtin_amdgcn_wmma_f32_16x16x32_bf16(
            false, a.v, false, b[j].v, (short)0, acc[mt][j], false, false);
    }
  }
}

// ---------------------------------------------------------------------------
// Fused per-row MLP: [BM,128] -> up(256) -> LN -> 3x(dense+SiLU) -> FIN cols.
// GATE=true: input = m * (rbf @ Wrbf^T) (edge branch). Else input = X rows.
// ---------------------------------------------------------------------------
template<bool GATE, int FIN>
__global__ __launch_bounds__(256) void fused_mlp_kernel(
    const float* __restrict__ X, const float* __restrict__ rbf,
    const float* __restrict__ Wrbf,
    const unsigned short* __restrict__ pUp,
    const float* __restrict__ ln_g, const float* __restrict__ ln_b,
    const unsigned short* __restrict__ pWd, const float* __restrict__ bd,
    const float* __restrict__ Wfin, float* __restrict__ out, int M) {
  __shared__ unsigned short Abuf[BM * LDA];  // bf16 activations (33.8 KB)
  __shared__ float redS[BM * 8];             // per-wave LN partial sums
  __shared__ float redQ[BM * 8];             // per-wave LN partial sumsq
  __shared__ float muL[BM];
  __shared__ float rsL[BM];
  __shared__ float WrbfL[EMB * 6];
  __shared__ float rbfL[BM * 6];

  const int tid  = threadIdx.x;
  const int lane = tid & 31;
  const int wave = tid >> 5;
  const int nt0  = wave * 2;            // 2 n-tiles per wave, all 4 m-tiles
  const int r0   = blockIdx.x * BM;

  // ---- stage 1: build input rows in Abuf (bf16, cols 0..127) ----
  if constexpr (GATE) {
    for (int i = tid; i < EMB * 6; i += 256) WrbfL[i] = Wrbf[i];
    for (int i = tid; i < BM * 6; i += 256) {
      int e = i / 6;
      rbfL[i] = (r0 + e < M) ? rbf[(size_t)(r0 + e) * 6 + (i % 6)] : 0.f;
    }
    __syncthreads();
    for (int i = 0; i < (BM * EMB) / 256; ++i) {
      int idx = i * 256 + tid;
      int e = idx >> 7, f = idx & 127;
      float v = 0.f;
      if (r0 + e < M) {
        float g = 0.f;
        #pragma unroll
        for (int r = 0; r < 6; ++r) g += rbfL[e * 6 + r] * WrbfL[f * 6 + r];
        v = X[(size_t)(r0 + e) * EMB + f] * g;
      }
      Abuf[e * LDA + f] = f2bf(v);
    }
  } else {
    for (int i = 0; i < (BM * EMB) / 256; ++i) {
      int idx = i * 256 + tid;
      int e = idx >> 7, f = idx & 127;
      float v = (r0 + e < M) ? X[(size_t)(r0 + e) * EMB + f] : 0.f;
      Abuf[e * LDA + f] = f2bf(v);
    }
  }
  __syncthreads();

  // ---- stage 2: up-projection 128 -> 256 (WMMA), C stays in registers ----
  v8f acc[4][2];
  #pragma unroll
  for (int mt = 0; mt < 4; ++mt)
    #pragma unroll
    for (int j = 0; j < 2; ++j)
      #pragma unroll
      for (int i = 0; i < 8; ++i) acc[mt][j][i] = 0.f;
  gemm_mt4<EMB / 32>(Abuf, lane, pUp, nt0, acc);

  // ---- stage 3: LayerNorm from fp32 accumulators ----
  // per-lane partials over this wave's 2 n-tiles, butterfly over lanes 0..15
  #pragma unroll
  for (int mt = 0; mt < 4; ++mt) {
    #pragma unroll
    for (int i = 0; i < 8; ++i) {
      float p = acc[mt][0][i] + acc[mt][1][i];
      float q = acc[mt][0][i] * acc[mt][0][i] + acc[mt][1][i] * acc[mt][1][i];
      #pragma unroll
      for (int msk = 8; msk >= 1; msk >>= 1) {   // reduce across lane&15
        p += __shfl_xor(p, msk, 32);
        q += __shfl_xor(q, msk, 32);
      }
      if ((lane & 15) == 0) {                    // lanes 0 and 16 hold row totals
        int row = mt * 16 + ((lane >> 4) << 3) + i;
        redS[row * 8 + wave] = p;
        redQ[row * 8 + wave] = q;
      }
    }
  }
  __syncthreads();   // also ensures all waves finished reading Abuf
  if (tid < BM) {
    float s = 0.f, q = 0.f;
    #pragma unroll
    for (int w = 0; w < 8; ++w) { s += redS[tid * 8 + w]; q += redQ[tid * 8 + w]; }
    float mu  = s * (1.f / D_HID);
    float var = q * (1.f / D_HID) - mu * mu;
    muL[tid] = mu;
    rsL[tid] = rsqrtf(var + LN_EPS);
  }
  __syncthreads();
  #pragma unroll
  for (int j = 0; j < 2; ++j) {
    int col = (nt0 + j) * 16 + (lane & 15);
    float g = ln_g[col], b = ln_b[col];
    #pragma unroll
    for (int mt = 0; mt < 4; ++mt) {
      #pragma unroll
      for (int i = 0; i < 8; ++i) {
        int row = mt * 16 + ((lane >> 4) << 3) + i;
        float y = (acc[mt][j][i] - muL[row]) * rsL[row] * g + b;
        Abuf[row * LDA + col] = f2bf(y);
      }
    }
  }
  __syncthreads();

  // ---- stage 4: three 256x256 dense layers + bias + SiLU ----
  for (int l = 0; l < 3; ++l) {
    #pragma unroll
    for (int j = 0; j < 2; ++j) {
      float b = bd[l * D_HID + (nt0 + j) * 16 + (lane & 15)];
      #pragma unroll
      for (int mt = 0; mt < 4; ++mt)
        #pragma unroll
        for (int i = 0; i < 8; ++i) acc[mt][j][i] = b;
    }
    gemm_mt4<D_HID / 32>(Abuf, lane, pWd + (size_t)l * D_HID * D_HID, nt0, acc);
    __syncthreads();   // all waves done reading Abuf before overwrite
    #pragma unroll
    for (int j = 0; j < 2; ++j) {
      int col = (nt0 + j) * 16 + (lane & 15);
      #pragma unroll
      for (int mt = 0; mt < 4; ++mt) {
        #pragma unroll
        for (int i = 0; i < 8; ++i) {
          int row = mt * 16 + ((lane >> 4) << 3) + i;
          float x = acc[mt][j][i];
          Abuf[row * LDA + col] = f2bf(x / (1.f + __expf(-x)));   // SiLU
        }
      }
    }
    __syncthreads();
  }

  // ---- stage 5: final fp32 projection 256 -> FIN ----
  for (int o = tid; o < BM * FIN; o += 256) {
    int row = o / FIN, col = o % FIN;
    if (r0 + row >= M) continue;
    const float* wr = Wfin + (size_t)col * D_HID;
    float a = 0.f;
    #pragma unroll 8
    for (int k = 0; k < D_HID; ++k) a += bf2f(Abuf[row * LDA + k]) * wr[k];
    out[(size_t)(r0 + row) * FIN + col] = a;
  }
}

// ---------------------------------------------------------------------------
extern "C" void kernel_launch(void* const* d_in, const int* in_sizes, int n_in,
                              void* d_out, int out_size, void* d_ws, size_t ws_size,
                              hipStream_t stream) {
  const float* m      = (const float*)d_in[0];
  const float* rbf    = (const float*)d_in[1];
  const int*   src    = (const int*)d_in[2];
  const float* Wrbf_e = (const float*)d_in[3];
  const float* Wup_e  = (const float*)d_in[4];
  const float* lng_e  = (const float*)d_in[5];
  const float* lnb_e  = (const float*)d_in[6];
  const float* Wd_e   = (const float*)d_in[7];
  const float* bd_e   = (const float*)d_in[8];
  const float* Wfin_e = (const float*)d_in[9];
  const float* Wrbf_n = (const float*)d_in[10];
  const float* Wup_n  = (const float*)d_in[11];
  const float* lng_n  = (const float*)d_in[12];
  const float* lnb_n  = (const float*)d_in[13];
  const float* Wd_n   = (const float*)d_in[14];
  const float* bd_n   = (const float*)d_in[15];
  const float* Wfin_n = (const float*)d_in[16];

  const int E = in_sizes[2];                       // 480000 edges
  const int N = (out_size - 2 * E) / 95;           // 30000 nodes

  // workspace layout
  char* ws = (char*)d_ws;
  size_t off = 0;
  auto alloc = [&](size_t bytes) { void* p = ws + off; off = (off + bytes + 255) & ~(size_t)255; return p; };
  float*          agg  = (float*)         alloc((size_t)N * EMB * sizeof(float));
  unsigned short* pUpE = (unsigned short*)alloc((size_t)D_HID * EMB * 2);
  unsigned short* pWdE = (unsigned short*)alloc((size_t)3 * D_HID * D_HID * 2);
  unsigned short* pUpN = (unsigned short*)alloc((size_t)D_HID * EMB * 2);
  unsigned short* pWdN = (unsigned short*)alloc((size_t)3 * D_HID * D_HID * 2);
  (void)ws_size;

  const int tilesUp = (D_HID / 16) * (EMB / 32);   // 64
  const int tilesD  = (D_HID / 16) * (D_HID / 32); // 128

  pack_w_kernel<<<tilesUp, 32, 0, stream>>>(Wup_e, pUpE, EMB);
  for (int l = 0; l < 3; ++l)
    pack_w_kernel<<<tilesD, 32, 0, stream>>>(Wd_e + (size_t)l * D_HID * D_HID,
                                             pWdE + (size_t)l * D_HID * D_HID, D_HID);
  pack_w_kernel<<<tilesUp, 32, 0, stream>>>(Wup_n, pUpN, EMB);
  for (int l = 0; l < 3; ++l)
    pack_w_kernel<<<tilesD, 32, 0, stream>>>(Wd_n + (size_t)l * D_HID * D_HID,
                                             pWdN + (size_t)l * D_HID * D_HID, D_HID);

  zero_kernel<<<(N * EMB + 255) / 256, 256, 0, stream>>>(agg, N * EMB);
  scatter_tn_kernel<<<(E * EMB + 255) / 256, 256, 0, stream>>>(m, rbf, src, Wrbf_n, agg, E * EMB);

  float* out_e = (float*)d_out;
  float* out_n = out_e + (size_t)E * 2;

  fused_mlp_kernel<true, 2><<<(E + BM - 1) / BM, 256, 0, stream>>>(
      m, rbf, Wrbf_e, pUpE, lng_e, lnb_e, pWdE, bd_e, Wfin_e, out_e, E);
  fused_mlp_kernel<false, 95><<<(N + BM - 1) / BM, 256, 0, stream>>>(
      agg, nullptr, nullptr, pUpN, lng_n, lnb_n, pWdN, bd_n, Wfin_n, out_n, N);
}
